// MalConvGCG_75144747811160
// MI455X (gfx1250) — matrix operand order repaired and checked by reading, not compiled, not run on hardware
//
#include <hip/hip_runtime.h>
#include <hip/hip_bf16.h>
#include <stdint.h>

typedef __attribute__((ext_vector_type(16))) __bf16 v16bf;
typedef __attribute__((ext_vector_type(8)))  __bf16 v8bf;
typedef __attribute__((ext_vector_type(8)))  float  v8f;

#define E_    8
#define C_    128
#define KW    512
#define B_    4
#define T_    1000000
#define TOUT  1953              /* (T - K)/S + 1 */
#define NWIN  (B_*TOUT)         /* 7812 */
#define KK    (E_*KW)           /* 4096 */
#define MT    ((NWIN + 15)/16)  /* 489  */
#define NT8   (C_/16)           /* 8    */

static __device__ __forceinline__ unsigned short f2bf(float f) {
  unsigned u = __float_as_uint(f);
  unsigned r = u + 0x7FFFu + ((u >> 16) & 1u);   // round-to-nearest-even
  return (unsigned short)(r >> 16);
}

static __device__ __forceinline__ v16bf cat8(v8bf lo, v8bf hi) {
  return __builtin_shufflevector(lo, hi, 0,1,2,3,4,5,6,7,8,9,10,11,12,13,14,15);
}

static __device__ __forceinline__ v16bf ldfrag16(const unsigned short* p0,
                                                 const unsigned short* p1) {
  v8bf lo = *(const v8bf*)p0;
  v8bf hi = *(const v8bf*)p1;
  return cat8(lo, hi);
}

#define WMMA_BF16(A, Bf, Cc) \
  __builtin_amdgcn_wmma_f32_16x16x32_bf16(false, (A), false, (Bf), (short)0, (Cc), false, false)

// ---------- prep kernels ----------------------------------------------------
__global__ void k_cvt_f32_bf16(const float* __restrict__ in,
                               unsigned short* __restrict__ out, int n) {
  for (int i = blockIdx.x*blockDim.x + threadIdx.x; i < n;
       i += gridDim.x*blockDim.x)
    out[i] = f2bf(in[i]);
}

// conv weight (2C, E, K) -> (2C, K*E) with element (o, k*8+e), bf16
__global__ void k_wtrans(const float* __restrict__ w,
                         unsigned short* __restrict__ wt) {
  int i = blockIdx.x*blockDim.x + threadIdx.x;
  if (i >= 2*C_*KK) return;
  int o = i >> 12;            // /4096
  int r = i & 4095;
  int e = r >> 9;             // /512
  int k = r & 511;
  wt[(size_t)o*KK + k*E_ + e] = f2bf(w[i]);
}

// ---------- stage 1: fused conv GEMM + GLU (both branches) ------------------
__global__ void __launch_bounds__(32)
k_conv_glu(const unsigned short* __restrict__ xbf,   // (B*T*E) bf16
           const unsigned short* __restrict__ wtC,   // (256, 4096) bf16
           const unsigned short* __restrict__ wtM,
           const float* __restrict__ biasC,          // (256)
           const float* __restrict__ biasM,
           unsigned short* __restrict__ Hc,          // (NWIN, 128) bf16
           unsigned short* __restrict__ Hm)
{
  const int mt   = blockIdx.x / NT8;
  const int nt   = blockIdx.x % NT8;
  const int lane = threadIdx.x & 31;
  const int lr   = lane & 15;
  const int kh   = lane >> 4;

  // A row (window) for this lane, clamped at edge tile
  int row = mt*16 + lr;
  int rc  = row < NWIN ? row : (NWIN-1);
  int b   = rc / TOUT;
  int t   = rc - b*TOUT;
  const unsigned short* a0 = xbf + (size_t)b*((size_t)T_*E_)
                                 + (size_t)t*KK + kh*16;

  const int ncol = nt*16 + lr;
  const unsigned short* bca = wtC + (size_t)ncol*KK        + kh*16;
  const unsigned short* bcb = wtC + (size_t)(ncol+C_)*KK   + kh*16;
  const unsigned short* bma = wtM + (size_t)ncol*KK        + kh*16;
  const unsigned short* bmb = wtM + (size_t)(ncol+C_)*KK   + kh*16;

  v8f acc_ca = {}; v8f acc_cb = {}; v8f acc_ma = {}; v8f acc_mb = {};

  // software pipeline, prefetch distance = 1 K-chunk (32)
  v16bf af  = ldfrag16(a0,  a0  + 8);
  v16bf fca = ldfrag16(bca, bca + 8);
  v16bf fcb = ldfrag16(bcb, bcb + 8);
  v16bf fm0 = ldfrag16(bma, bma + 8);
  v16bf fm1 = ldfrag16(bmb, bmb + 8);

  for (int kc = 0; kc < KK - 32; kc += 32) {
    const int kn = kc + 32;
    v16bf af_n  = ldfrag16(a0 + kn,  a0 + kn + 8);
    v16bf fca_n = ldfrag16(bca + kn, bca + kn + 8);
    v16bf fcb_n = ldfrag16(bcb + kn, bcb + kn + 8);
    v16bf fm0_n = ldfrag16(bma + kn, bma + kn + 8);
    v16bf fm1_n = ldfrag16(bmb + kn, bmb + kn + 8);

    acc_ca = WMMA_BF16(af, fca, acc_ca);
    acc_cb = WMMA_BF16(af, fcb, acc_cb);
    acc_ma = WMMA_BF16(af, fm0, acc_ma);
    acc_mb = WMMA_BF16(af, fm1, acc_mb);

    af = af_n; fca = fca_n; fcb = fcb_n; fm0 = fm0_n; fm1 = fm1_n;
  }
  acc_ca = WMMA_BF16(af, fca, acc_ca);
  acc_cb = WMMA_BF16(af, fcb, acc_cb);
  acc_ma = WMMA_BF16(af, fm0, acc_ma);
  acc_mb = WMMA_BF16(af, fm1, acc_mb);

  const float bCa = biasC[ncol], bCb = biasC[ncol + C_];
  const float bMa = biasM[ncol], bMb = biasM[ncol + C_];
  #pragma unroll
  for (int i = 0; i < 8; ++i) {
    int m = mt*16 + i + kh*8;
    if (m < NWIN) {
      float a = acc_ca[i] + bCa;
      float g = acc_cb[i] + bCb;
      Hc[(size_t)m*C_ + ncol] = f2bf(a * (1.0f/(1.0f + __expf(-g))));
      a = acc_ma[i] + bMa;
      g = acc_mb[i] + bMb;
      Hm[(size_t)m*C_ + ncol] = f2bf(a * (1.0f/(1.0f + __expf(-g))));
    }
  }
}

// ---------- stage 2: 1x1 share conv (K=128 GEMM) + leaky --------------------
__global__ void __launch_bounds__(32)
k_share(const unsigned short* __restrict__ H,    // (NWIN,128) bf16
        const unsigned short* __restrict__ Wbf,  // (128,128) bf16, row o
        const float* __restrict__ bias,
        float* __restrict__ G)                   // (NWIN,128) f32
{
  const int mt   = blockIdx.x / NT8;
  const int nt   = blockIdx.x % NT8;
  const int lane = threadIdx.x & 31;
  const int lr   = lane & 15;
  const int kh   = lane >> 4;

  int row = mt*16 + lr;
  int rc  = row < NWIN ? row : (NWIN-1);
  const unsigned short* a0 = H + (size_t)rc*C_ + kh*16;
  const int ncol = nt*16 + lr;
  const unsigned short* b0 = Wbf + (size_t)ncol*C_ + kh*16;

  v8f acc = {};
  #pragma unroll
  for (int kc = 0; kc < C_; kc += 32) {
    v16bf af  = ldfrag16(a0 + kc, a0 + kc + 8);
    v16bf bf_ = ldfrag16(b0 + kc, b0 + kc + 8);
    acc = WMMA_BF16(af, bf_, acc);
  }
  const float bo = bias[ncol];
  #pragma unroll
  for (int i = 0; i < 8; ++i) {
    int m = mt*16 + i + kh*8;
    if (m < NWIN) {
      float v = acc[i] + bo;
      G[(size_t)m*C_ + ncol] = v >= 0.0f ? v : 0.01f*v;
    }
  }
}

// ---------- stage 3..6: reductions & gating ---------------------------------
__global__ void k_ctx_max(const float* __restrict__ Gc, float* __restrict__ gct) {
  int b = blockIdx.x, c = threadIdx.x;
  const float* p = Gc + (size_t)b*TOUT*C_ + c;
  float mx = -3.402823466e38f;
  for (int t = 0; t < TOUT; ++t) mx = fmaxf(mx, p[(size_t)t*C_]);
  gct[b*C_ + c] = mx;
}

__global__ void k_q(const float* __restrict__ gct, const float* __restrict__ Wq,
                    const float* __restrict__ bq, float* __restrict__ q) {
  int o = threadIdx.x;
  for (int b = 0; b < B_; ++b) {
    float s = bq[o];
    for (int c = 0; c < C_; ++c) s += gct[b*C_ + c] * Wq[o*C_ + c];
    q[b*C_ + o] = tanhf(s);
  }
}

__global__ void __launch_bounds__(256)
k_gate(const float* __restrict__ Gm, const float* __restrict__ q,
       float* __restrict__ gate) {
  int wid  = (blockIdx.x*blockDim.x + threadIdx.x) >> 5;
  int lane = threadIdx.x & 31;
  if (wid >= NWIN) return;
  int b = wid / TOUT;
  const float* r  = Gm + (size_t)wid*C_;
  const float* qq = q + b*C_;
  float s = 0.0f;
  #pragma unroll
  for (int j = 0; j < C_/32; ++j) s += r[lane + j*32] * qq[lane + j*32];
  for (int off = 16; off; off >>= 1) s += __shfl_xor(s, off, 32);
  if (lane == 0) gate[wid] = 1.0f/(1.0f + __expf(-s));
}

__global__ void k_out(const float* __restrict__ Gm, const float* __restrict__ gate,
                      float* __restrict__ out) {
  int b = blockIdx.x, c = threadIdx.x;
  const float* p  = Gm + (size_t)b*TOUT*C_ + c;
  const float* gp = gate + b*TOUT;
  float mx = -3.402823466e38f;
  for (int t = 0; t < TOUT; ++t) mx = fmaxf(mx, p[(size_t)t*C_] * gp[t]);
  out[b*C_ + c] = mx;
}

// ---------- host launcher ---------------------------------------------------
extern "C" void kernel_launch(void* const* d_in, const int* in_sizes, int n_in,
                              void* d_out, int out_size, void* d_ws, size_t ws_size,
                              hipStream_t stream) {
  const float* x      = (const float*)d_in[0];
  const float* wcC    = (const float*)d_in[1];
  const float* bcC    = (const float*)d_in[2];
  const float* wsC_f  = (const float*)d_in[3];
  const float* bsC    = (const float*)d_in[4];
  const float* wcM    = (const float*)d_in[5];
  const float* bcM    = (const float*)d_in[6];
  const float* wsM_f  = (const float*)d_in[7];
  const float* bsM    = (const float*)d_in[8];
  const float* wq     = (const float*)d_in[9];
  const float* bq     = (const float*)d_in[10];
  float* out = (float*)d_out;

  // workspace carve (all 256B aligned)
  size_t off = 0;
  char* base = (char*)d_ws;
  auto carve = [&](size_t bytes) {
    void* p = base + off;
    off += (bytes + 255) & ~(size_t)255;
    return p;
  };
  unsigned short* xbf = (unsigned short*)carve((size_t)B_*T_*E_*2);   // 64 MB
  unsigned short* wtC = (unsigned short*)carve((size_t)2*C_*KK*2);
  unsigned short* wtM = (unsigned short*)carve((size_t)2*C_*KK*2);
  unsigned short* sbC = (unsigned short*)carve((size_t)C_*C_*2);
  unsigned short* sbM = (unsigned short*)carve((size_t)C_*C_*2);
  unsigned short* Hc  = (unsigned short*)carve((size_t)NWIN*C_*2);
  unsigned short* Hm  = (unsigned short*)carve((size_t)NWIN*C_*2);
  float* Gc   = (float*)carve((size_t)NWIN*C_*4);
  float* Gm   = (float*)carve((size_t)NWIN*C_*4);
  float* gct  = (float*)carve((size_t)B_*C_*4);
  float* q    = (float*)carve((size_t)B_*C_*4);
  float* gate = (float*)carve((size_t)NWIN*4);
  (void)ws_size; (void)in_sizes; (void)n_in; (void)out_size;

  // prep: convert x, share weights; transpose conv weights
  k_cvt_f32_bf16<<<16384, 256, 0, stream>>>(x, xbf, B_*T_*E_);
  k_cvt_f32_bf16<<<64, 256, 0, stream>>>(wsC_f, sbC, C_*C_);
  k_cvt_f32_bf16<<<64, 256, 0, stream>>>(wsM_f, sbM, C_*C_);
  {
    int n = 2*C_*KK;
    k_wtrans<<<(n + 255)/256, 256, 0, stream>>>(wcC, wtC);
    k_wtrans<<<(n + 255)/256, 256, 0, stream>>>(wcM, wtM);
  }

  // stage 1: conv GEMM + GLU (both branches, 4 WMMA per A-fragment)
  k_conv_glu<<<MT*NT8, 32, 0, stream>>>(xbf, wtC, wtM, bcC, bcM, Hc, Hm);

  // stage 2: share 1x1 conv + leaky
  k_share<<<MT*NT8, 32, 0, stream>>>(Hc, sbC, bsC, Gc);
  k_share<<<MT*NT8, 32, 0, stream>>>(Hm, sbM, bsM, Gm);

  // stage 3..6
  k_ctx_max<<<B_, C_, 0, stream>>>(Gc, gct);
  k_q<<<1, C_, 0, stream>>>(gct, wq, bq, q);
  k_gate<<<(NWIN + 7)/8, 256, 0, stream>>>(Gm, q, gate);
  k_out<<<B_, C_, 0, stream>>>(Gm, gate, out);
}
